// MLAAttention_13486197310129
// MI455X (gfx1250) — matrix operand order repaired and checked by reading, not compiled, not run on hardware
//
#include <hip/hip_runtime.h>
#include <cstdint>

// ---------------- types / constants ----------------
typedef __attribute__((ext_vector_type(16))) __bf16 v16bf;
typedef __attribute__((ext_vector_type(8)))  float  v8f;
typedef unsigned int u32x4 __attribute__((ext_vector_type(4)));
typedef int          i32x8 __attribute__((ext_vector_type(8)));
typedef int          i32x4 __attribute__((ext_vector_type(4)));

#define S_LEN 2048
#define HID   2048
#define NH    16
#define DQK   192       // 128 nope + 64 rope
#define DV    128
#define QLAT  768
#define KVLAT 512
#define QK_SCALE 0.07216878364870322f   // 1/sqrt(192)
#define NEGBIG (-1e9f)

union BF8 { float4 f4; __bf16 h[8]; };

// Build a 16-elem bf16 fragment from two 8-elem (16B) loads.
static __device__ inline v16bf ld16(const __bf16* p0, const __bf16* p1) {
  BF8 a, b;
  a.f4 = *(const float4*)p0;
  b.f4 = *(const float4*)p1;
  v16bf r;
  #pragma unroll
  for (int j = 0; j < 8; ++j) { r[j] = a.h[j]; r[8 + j] = b.h[j]; }
  return r;
}

// ---------------- Tensor Data Mover: 2D tile global -> LDS ----------------
// D# per CDNA5 ISA: group0 {count, lds_addr,
// global_addr, type=2}; group1 {data_size=2B, tensor_dim0/1, tile_dim0/1,
// tensor_dim0_stride}; groups 2/3 zero (<=2D tensor).
// This toolchain's builtin is the 6-arg form (extra int32x8 group + cpol).
static __device__ inline void tdm_load_2d(uint32_t lds_off, const void* gptr,
                                          int width, int height, int row_stride_elems) {
  unsigned long long ga = (unsigned long long)(uintptr_t)gptr;
  u32x4 g0;
  g0.x = 1u;                                                  // count = 1
  g0.y = lds_off;                                             // lds_addr (bytes)
  g0.z = (unsigned)(ga & 0xFFFFFFFFu);                        // global_addr[31:0]
  g0.w = (unsigned)((ga >> 32) & 0x1FFFFFFu) | (2u << 30);    // addr[56:32] | type=2
  i32x8 g1;
  g1[0] = 1 << 16;                                            // data_size=1 -> 2 bytes
  g1[1] = (width & 0xFFFF) << 16;                             // tensor_dim0[15:0]
  g1[2] = ((width >> 16) & 0xFFFF) | ((height & 0xFFFF) << 16);   // dim0 hi | tensor_dim1 lo
  g1[3] = ((height >> 16) & 0xFFFF) | ((width & 0xFFFF) << 16);   // dim1 hi | tile_dim0
  g1[4] = height & 0xFFFF;                                    // tile_dim1 (tile_dim2=0)
  g1[5] = row_stride_elems;                                   // tensor_dim0_stride[31:0]
  g1[6] = 0;
  g1[7] = 0;
  i32x4 z4 = {0, 0, 0, 0};
  i32x8 z8 = {0, 0, 0, 0, 0, 0, 0, 0};
  __builtin_amdgcn_tensor_load_to_lds(g0, g1, z4, z4, z8, 0);
}

// ---------------- fp32 -> bf16 convert (plain) ----------------
__global__ void k_cvt(const float* __restrict__ in, __bf16* __restrict__ out, int n) {
  int i = blockIdx.x * 256 + threadIdx.x;
  if (i < n) out[i] = (__bf16)in[i];
}

// fp32 [K,N] -> bf16 transposed [N,K] (weights become K-contiguous per column)
__global__ void k_cvt_t(const float* __restrict__ in, __bf16* __restrict__ out,
                        int Kdim, int Ndim) {
  int i = blockIdx.x * 256 + threadIdx.x;
  if (i >= Kdim * Ndim) return;
  int n = i / Kdim, k = i % Kdim;
  out[i] = (__bf16)in[(size_t)k * Ndim + n];
}

// Fuse [K,H*128] nope-up and [K,H*64] rope weights into TRANSPOSED [H*192, K] bf16.
__global__ void k_fuse_t(const float* __restrict__ Wup, const float* __restrict__ Wrope,
                         __bf16* __restrict__ out, int Kdim) {
  int i = blockIdx.x * 256 + threadIdx.x;
  if (i >= NH * DQK * Kdim) return;
  int c = i / Kdim, k = i % Kdim;
  int h = c / DQK, d = c % DQK;
  float v = (d < 128) ? Wup[(size_t)k * (NH * 128) + h * 128 + d]
                      : Wrope[(size_t)k * (NH * 64) + h * 64 + (d - 128)];
  out[i] = (__bf16)v;
}

// v [B*S, H*DV] -> vT [(b,h,d), s]  (column-major per head, for vectorized PV frags)
__global__ void k_tr_v(const __bf16* __restrict__ in, __bf16* __restrict__ out) {
  int i = blockIdx.x * 256 + threadIdx.x;
  if (i >= 2 * NH * DV * S_LEN) return;
  int s = i & (S_LEN - 1);
  int d = (i >> 11) & (DV - 1);
  int h = (i >> 18) & (NH - 1);
  int b = i >> 22;
  out[i] = in[(size_t)(b * S_LEN + s) * (NH * DV) + h * DV + d];
}

// ---------------- decoupled RoPE (in place on fused q/k) ----------------
__global__ void k_rope(__bf16* __restrict__ Qf, __bf16* __restrict__ Kf,
                       const int* __restrict__ pos) {
  int gid = blockIdx.x * 256 + threadIdx.x;     // B*S*H*32 threads
  if (gid >= 2 * S_LEN * NH * 32) return;
  int i = gid & 31;                              // rope pair (i, i+32)
  int bsh = gid >> 5;
  int h = bsh & (NH - 1);
  int bs = bsh >> 4;
  int s = bs & (S_LEN - 1);
  int b = bs >> 11;
  float t = (float)pos[b * S_LEN + s];
  float inv_freq = __powf(10000.0f, -(float)i / 32.0f);
  float ang = t * inv_freq;
  float c = __cosf(ang), sn = __sinf(ang);
  size_t off = (size_t)bs * (NH * DQK) + h * DQK + 128;
  {
    float x1 = (float)Qf[off + i], x2 = (float)Qf[off + i + 32];
    Qf[off + i]      = (__bf16)(x1 * c - x2 * sn);
    Qf[off + i + 32] = (__bf16)(x2 * c + x1 * sn);
  }
  {
    float x1 = (float)Kf[off + i], x2 = (float)Kf[off + i + 32];
    Kf[off + i]      = (__bf16)(x1 * c - x2 * sn);
    Kf[off + i + 32] = (__bf16)(x2 * c + x1 * sn);
  }
}

// ---------------- LDS-tiled bf16 WMMA GEMM: C[M,N] = A[M,K] * BT[N,K]^T ----------------
// Requires M%64==0, N%64==0, K%32==0.  128 threads: 4 waves, each 32x32.
// Both A and BT tiles are K-contiguous -> all fragments are 2x ds_load_b128.
template <typename OutT>
__global__ __launch_bounds__(128) void gemm_bf16t(
    const __bf16* __restrict__ A, const __bf16* __restrict__ BT,
    OutT* __restrict__ C, int M, int N, int K)
{
  __shared__ __align__(16) __bf16 As[64][32];
  __shared__ __align__(16) __bf16 Bst[64][32];
  const int tid = threadIdx.x;
  const int lane = tid & 31, wv = tid >> 5;
  const int half = lane >> 4, ln = lane & 15;
  const int bm = blockIdx.y * 64, bn = blockIdx.x * 64;
  const int wrow = (wv >> 1) * 32, wcol = (wv & 1) * 32;

  v8f acc[2][2] = {};

  for (int k0 = 0; k0 < K; k0 += 32) {
    __syncthreads();
    {
      int r = tid >> 1, c = (tid & 1) * 16;
      const __bf16* sa = A + (size_t)(bm + r) * K + k0 + c;
      *(float4*)&As[r][c]      = *(const float4*)sa;
      *(float4*)&As[r][c + 8]  = *(const float4*)(sa + 8);
      const __bf16* sb = BT + (size_t)(bn + r) * K + k0 + c;
      *(float4*)&Bst[r][c]     = *(const float4*)sb;
      *(float4*)&Bst[r][c + 8] = *(const float4*)(sb + 8);
      if (k0 + 32 < K) {                       // prefetch next K tile
        __builtin_prefetch(sa + 32, 0, 0);
        __builtin_prefetch(sb + 32, 0, 0);
      }
    }
    __syncthreads();

    v16bf af[2], bfv[2];
    #pragma unroll
    for (int ti = 0; ti < 2; ++ti) {
      const __bf16* p = &As[wrow + ti * 16 + ln][half * 16];
      af[ti] = ld16(p, p + 8);                // A layout: K=(half*16)+j at row ln
    }
    #pragma unroll
    for (int tj = 0; tj < 2; ++tj) {
      const __bf16* p = &Bst[wcol + tj * 16 + ln][half * 16];
      bfv[tj] = ld16(p, p + 8);               // B layout: K=(half*16)+j at col ln
    }
    #pragma unroll
    for (int ti = 0; ti < 2; ++ti)
      #pragma unroll
      for (int tj = 0; tj < 2; ++tj)
        acc[ti][tj] = __builtin_amdgcn_wmma_f32_16x16x32_bf16(
            false, af[ti], false, bfv[tj], (short)0, acc[ti][tj], false, false);
  }

  #pragma unroll
  for (int ti = 0; ti < 2; ++ti)
    #pragma unroll
    for (int tj = 0; tj < 2; ++tj)
      #pragma unroll
      for (int v = 0; v < 8; ++v) {
        int row = bm + wrow + ti * 16 + v + 8 * half;   // C layout
        int col = bn + wcol + tj * 16 + ln;
        C[(size_t)row * N + col] = (OutT)acc[ti][tj][v];
      }
}

// ---------------- online-softmax update, specialized on diagonal masking ----------------
// Score C layout: VGPR v holds row (v + 8*half), lane column = nt*16 + ln.
template <bool MASKED>
static __device__ inline void softmax_update(
    const v8f (&sacc)[4], v8f (&oacc)[8], float (&mrow)[8], float (&lrow)[8],
    __bf16 (*Psw)[64], int qrow0, int kcol0, int half, int ln)
{
  #pragma unroll
  for (int v = 0; v < 8; ++v) {
    const int qq = qrow0 + v + 8 * half;
    float sv[4];
    float rmax = -INFINITY;
    #pragma unroll
    for (int nt = 0; nt < 4; ++nt) {
      float s = sacc[nt][v] * QK_SCALE;
      if (MASKED) {
        int kk = kcol0 + nt * 16 + ln;
        if (kk > qq) s += NEGBIG;
      }
      sv[nt] = s;
      rmax = fmaxf(rmax, s);
    }
    #pragma unroll
    for (int o = 1; o < 16; o <<= 1) rmax = fmaxf(rmax, __shfl_xor(rmax, o, 32));
    float mnew = fmaxf(mrow[v], rmax);
    float corr = __expf(mrow[v] - mnew);
    float psum = 0.0f;
    #pragma unroll
    for (int nt = 0; nt < 4; ++nt) {
      float p = __expf(sv[nt] - mnew);
      psum += p;
      Psw[v + 8 * half][nt * 16 + ln] = (__bf16)p;     // C layout -> LDS
    }
    #pragma unroll
    for (int o = 1; o < 16; o <<= 1) psum += __shfl_xor(psum, o, 32);
    lrow[v] = lrow[v] * corr + psum;
    mrow[v] = mnew;
    #pragma unroll
    for (int j = 0; j < 8; ++j) oacc[j][v] = oacc[j][v] * corr;
  }
}

// ---------------- causal flash attention: TDM tiles + bf16 WMMA ----------------
// grid: (S/64, NH, B), block 128.  Wave w owns q rows [qt*64+w*16, +16).
__global__ __launch_bounds__(128) void flash_mla(
    const __bf16* __restrict__ Q, const __bf16* __restrict__ K,
    const __bf16* __restrict__ VT, __bf16* __restrict__ O)
{
  __shared__ __align__(16) __bf16 Ks[64][DQK];    // 24 KB, row-major [kv][d]
  __shared__ __align__(16) __bf16 Vst[DV][64];    // 16 KB, [d][kv] (from vT)
  __shared__ __align__(16) __bf16 Ps[4][16][64];  //  8 KB, per-wave P staging

  const int qt = blockIdx.x, h = blockIdx.y, b = blockIdx.z;
  const int tid = threadIdx.x;
  const int lane = tid & 31, wv = tid >> 5;
  const int half = lane >> 4, ln = lane & 15;

  const __bf16* kbase  = K  + ((size_t)(b * S_LEN) * NH + h) * DQK;
  const __bf16* vtbase = VT + (size_t)(b * NH + h) * DV * S_LEN;
  const uint32_t ks_off  = (uint32_t)(uintptr_t)&Ks[0][0];
  const uint32_t vst_off = (uint32_t)(uintptr_t)&Vst[0][0];

  // Q fragments resident in registers (16 rows x 192, 6 K-steps of 32)
  v16bf qf[6];
  {
    const int r = qt * 64 + wv * 16 + ln;
    const __bf16* qp = Q + ((size_t)(b * S_LEN + r) * NH + h) * DQK;
    #pragma unroll
    for (int kd = 0; kd < 6; ++kd) {
      const __bf16* p = qp + kd * 32 + half * 16;
      qf[kd] = ld16(p, p + 8);
    }
  }

  v8f oacc[8] = {};
  float mrow[8], lrow[8];
  #pragma unroll
  for (int v = 0; v < 8; ++v) { mrow[v] = -INFINITY; lrow[v] = 0.0f; }

  for (int kt = 0; kt <= qt; ++kt) {
    __syncthreads();                      // prior iteration's LDS reads done
    if (wv == 0) {                        // one TDM DMA per tile (EXEC ignored)
      tdm_load_2d(ks_off,  kbase + (size_t)(kt * 64) * (NH * DQK), DQK, 64, NH * DQK);
      tdm_load_2d(vst_off, vtbase + kt * 64,                       64,  DV, S_LEN);
      __builtin_amdgcn_s_wait_tensorcnt(0);
    }
    __syncthreads();                      // tiles visible to all waves

    // S = Q K^T : 16x64 fp32.  B[k_dim][n] = Ks[n][k_dim] (contiguous in j).
    v8f sacc[4] = {};
    #pragma unroll
    for (int kd = 0; kd < 6; ++kd) {
      #pragma unroll
      for (int nt = 0; nt < 4; ++nt) {
        const __bf16* p = &Ks[nt * 16 + ln][kd * 32 + half * 16];
        v16bf bfv = ld16(p, p + 8);
        sacc[nt] = __builtin_amdgcn_wmma_f32_16x16x32_bf16(
            false, qf[kd], false, bfv, (short)0, sacc[nt], false, false);
      }
    }

    // online softmax: causal masking only possible on the diagonal tile
    if (kt == qt)
      softmax_update<true >(sacc, oacc, mrow, lrow, Ps[wv],
                            qt * 64 + wv * 16, kt * 64, half, ln);
    else
      softmax_update<false>(sacc, oacc, mrow, lrow, Ps[wv],
                            qt * 64 + wv * 16, kt * 64, half, ln);

    // O += P(16x64) @ V(64x128).  Same-wave LDS is in-order -> Ps readable.
    // B frag from Vst[d][kv]: contiguous in k -> vectorized.
    #pragma unroll
    for (int kk0 = 0; kk0 < 2; ++kk0) {
      const __bf16* pp = &Ps[wv][ln][kk0 * 32 + half * 16];
      v16bf pf = ld16(pp, pp + 8);                        // A layout
      #pragma unroll
      for (int nt = 0; nt < 8; ++nt) {
        const __bf16* p = &Vst[nt * 16 + ln][kk0 * 32 + half * 16];
        v16bf vf = ld16(p, p + 8);                        // B layout
        oacc[nt] = __builtin_amdgcn_wmma_f32_16x16x32_bf16(
            false, pf, false, vf, (short)0, oacc[nt], false, false);
      }
    }
  }

  // epilogue: normalize, store bf16 [B*S, H*DV]
  {
    const int rbase = qt * 64 + wv * 16;
    #pragma unroll
    for (int v = 0; v < 8; ++v) {
      int m = v + 8 * half;
      float invl = 1.0f / lrow[v];
      __bf16* op = O + (size_t)(b * S_LEN + rbase + m) * (NH * DV) + h * DV;
      #pragma unroll
      for (int nt = 0; nt < 8; ++nt)
        op[nt * 16 + ln] = (__bf16)(oacc[nt][v] * invl);
    }
  }
}

// ---------------- launch ----------------
extern "C" void kernel_launch(void* const* d_in, const int* in_sizes, int n_in,
                              void* d_out, int out_size, void* d_ws, size_t ws_size,
                              hipStream_t stream)
{
  const float* hs   = (const float*)d_in[0];
  // d_in[1] attention_mask: causal additive mask, implemented analytically
  const float* Wqd  = (const float*)d_in[2];
  const float* Wqu  = (const float*)d_in[3];
  const float* Wqr  = (const float*)d_in[4];
  const float* Wkvd = (const float*)d_in[5];
  const float* Wku  = (const float*)d_in[6];
  const float* Wkr  = (const float*)d_in[7];
  const float* Wvu  = (const float*)d_in[8];
  const float* Wo   = (const float*)d_in[9];
  const int*   pos  = (const int*)d_in[10];
  float* out = (float*)d_out;

  // workspace carve-up (bf16 elements), ~152 MB total
  __bf16* wsp = (__bf16*)d_ws;
  size_t off = 0;
  auto take = [&](size_t n) { __bf16* p = wsp + off; off += (n + 127) & ~(size_t)127; return p; };

  const size_t BS = (size_t)2 * S_LEN;                // 4096 token rows
  __bf16* hsb   = take(BS * HID);
  __bf16* wqdT  = take((size_t)HID * QLAT);           // [QLAT, HID]
  __bf16* wkvdT = take((size_t)HID * KVLAT);          // [KVLAT, HID]
  __bf16* wqfT  = take((size_t)QLAT * NH * DQK);      // [H*192, QLAT]
  __bf16* wkfT  = take((size_t)KVLAT * NH * DQK);     // [H*192, KVLAT]
  __bf16* wvuT  = take((size_t)KVLAT * NH * DV);      // [H*128, KVLAT]
  __bf16* woT   = take((size_t)NH * DV * HID);        // [HID, H*128]
  __bf16* qlat  = take(BS * QLAT);
  __bf16* ckv   = take(BS * KVLAT);
  __bf16* qfu   = take(BS * NH * DQK);
  __bf16* kfu   = take(BS * NH * DQK);
  __bf16* vfu   = take(BS * NH * DV);
  __bf16* vT    = take(BS * NH * DV);                 // [(b,h,d), s]
  __bf16* aout  = take(BS * NH * DV);

  k_cvt<<<dim3((unsigned)((BS * HID + 255) / 256)), 256, 0, stream>>>(hs, hsb, (int)(BS * HID));
  k_cvt_t<<<dim3((HID * QLAT + 255) / 256), 256, 0, stream>>>(Wqd, wqdT, HID, QLAT);
  k_cvt_t<<<dim3((HID * KVLAT + 255) / 256), 256, 0, stream>>>(Wkvd, wkvdT, HID, KVLAT);
  k_cvt_t<<<dim3((KVLAT * NH * DV + 255) / 256), 256, 0, stream>>>(Wvu, wvuT, KVLAT, NH * DV);
  k_cvt_t<<<dim3((NH * DV * HID + 255) / 256), 256, 0, stream>>>(Wo, woT, NH * DV, HID);
  k_fuse_t<<<dim3((QLAT * NH * DQK + 255) / 256), 256, 0, stream>>>(Wqu, Wqr, wqfT, QLAT);
  k_fuse_t<<<dim3((KVLAT * NH * DQK + 255) / 256), 256, 0, stream>>>(Wku, Wkr, wkfT, KVLAT);

  auto gemm_b = [&](const __bf16* A, const __bf16* BT, __bf16* C, int M, int N, int K) {
    gemm_bf16t<__bf16><<<dim3(N / 64, M / 64), 128, 0, stream>>>(A, BT, C, M, N, K);
  };
  gemm_b(hsb,  wqdT,  qlat, (int)BS, QLAT,     HID);   // q latent
  gemm_b(hsb,  wkvdT, ckv,  (int)BS, KVLAT,    HID);   // kv latent
  gemm_b(qlat, wqfT,  qfu,  (int)BS, NH * DQK, QLAT);  // fused q (nope|rope)
  gemm_b(ckv,  wkfT,  kfu,  (int)BS, NH * DQK, KVLAT); // fused k (nope|rope)
  gemm_b(ckv,  wvuT,  vfu,  (int)BS, NH * DV,  KVLAT); // v

  k_rope<<<dim3((2 * S_LEN * NH * 32) / 256), 256, 0, stream>>>(qfu, kfu, pos);
  k_tr_v<<<dim3((2 * NH * DV * S_LEN + 255) / 256), 256, 0, stream>>>(vfu, vT);

  flash_mla<<<dim3(S_LEN / 64, NH, 2), 128, 0, stream>>>(qfu, kfu, vT, aout);

  gemm_bf16t<float><<<dim3(HID / 64, (int)BS / 64), 128, 0, stream>>>(
      aout, woT, out, (int)BS, HID, NH * DV);
}
